// CustomGAT_46033459478728
// MI455X (gfx1250) — matrix operand order repaired and checked by reading, not compiled, not run on hardware
//
#include <hip/hip_runtime.h>
#include <hip/hip_bf16.h>

typedef __attribute__((ext_vector_type(16))) __bf16 v16bf;
typedef __attribute__((ext_vector_type(8)))  float  v8f;
typedef __attribute__((ext_vector_type(4)))  unsigned int v4ui;
typedef __attribute__((ext_vector_type(8)))  int    v8si;
typedef __attribute__((ext_vector_type(4)))  int    v4si;

#define HID 128
#define NT  8          // HID / 16 column tiles
#define NEG_SLOPE 0.2f

#if defined(__HIP_DEVICE_COMPILE__) && __has_builtin(__builtin_amdgcn_tensor_load_to_lds) && __has_builtin(__builtin_amdgcn_s_wait_tensorcnt)
#define GAT_HAVE_TDM 1
#else
#define GAT_HAVE_TDM 0
#endif

// ---------- helpers: monotonic float<->uint for atomicMax on f32 ----------
__device__ __forceinline__ unsigned f2mono(float f) {
    unsigned u = __float_as_uint(f);
    return (u & 0x80000000u) ? ~u : (u | 0x80000000u);
}
__device__ __forceinline__ float mono2f(unsigned u) {
    return (u & 0x80000000u) ? __uint_as_float(u & 0x7FFFFFFFu) : __uint_as_float(~u);
}

// ---------- weight pre-swizzle: f32 row-major [K, 128] -> WMMA bf16 B-fragments ----------
// Tile (kt, nt): lane L holds column nt*16 + (L&15), K-range kt*32 + (L>>4)*16 + [0..15],
// stored as 16 contiguous bf16 (32 B) per lane.
__global__ void gat_prep_b(const float* __restrict__ W, __bf16* __restrict__ out) {
    const int tile = blockIdx.x;           // (K/32)*NT tiles
    const int kt = tile / NT, nt = tile % NT;
    const int lane = threadIdx.x;          // 32 lanes
    const int col = nt * 16 + (lane & 15);
    const int kbase = kt * 32 + ((lane >> 4) << 4);
    __bf16* dst = out + ((size_t)tile * 32 + lane) * 16;
#pragma unroll
    for (int i = 0; i < 16; ++i)
        dst[i] = (__bf16)W[(size_t)(kbase + i) * HID + col];
}

// ---------- GEMM: C[N,128] = act(A[N,K] @ B + bias) via v_wmma_f32_16x16x32_bf16 ----------
// Block = 4 waves, 64 rows. B-fragments staged to LDS once per block via the
// Tensor Data Mover (tensor_load_to_lds, TENSORcnt), then fed from ds_load_b128.
__global__ __launch_bounds__(128)
void gat_gemm_wmma(const float* __restrict__ A, const __bf16* __restrict__ Bt,
                   const float* __restrict__ bias, float* __restrict__ C,
                   int nrows, int K, int flags /*1=bias, 2=relu*/) {
    const int lane = threadIdx.x & 31;
    const int wave = threadIdx.x >> 5;
    const int rowbase = blockIdx.x * 64 + wave * 16;
    const int half = lane >> 4;
    const int l15  = lane & 15;
    const int ktiles = K >> 5;

#if GAT_HAVE_TDM
    // 32 KB staging buffer: ktiles * NT tiles * 32 lanes * 32 B  (max ktiles = 4)
    __shared__ __align__(128) unsigned char ldsB[4 * NT * 32 * 32];
    // Flat pointer into LDS: addr[31:0] is the wave-relative LDS byte offset (ISA aperture rule)
    const unsigned ldsOff = (unsigned)(uintptr_t)(void*)ldsB;
    {
        const unsigned ndw = (unsigned)(ktiles * NT * 32 * 32) >> 2;   // dwords to move
        if (wave == 0) {
            const unsigned long long ga = (unsigned long long)Bt;      // tile start
            // D# group 0: count=1 | lds_addr | global_addr | type=2
            v4ui g0 = { 1u, ldsOff,
                        (unsigned)ga,
                        (unsigned)((ga >> 32) & 0x1FFFFFFu) | 0x80000000u };
            // D# group 1: data_size=4B; tensor_dim0 = tile_dim0 = ndw; tensor_dim1 = 1
            v8si g1 = { (int)(2u << 16),                 // data_size = 4 bytes
                        (int)((ndw & 0xFFFFu) << 16),    // tensor_dim0[15:0]
                        (int)(((ndw >> 16) & 0xFFFFu) | (1u << 16)), // dim0[31:16] | dim1=1
                        (int)((ndw & 0xFFFFu) << 16),    // tile_dim0 = ndw
                        1,                               // tile_dim1 = 1, tile_dim2 = 0
                        (int)ndw,                        // tensor_dim0_stride lo32
                        0, 0 };
            v4si gz4 = { 0, 0, 0, 0 };                   // groups 2/3 unused (2-D tensor)
            v8si gz8 = { 0, 0, 0, 0, 0, 0, 0, 0 };
            __builtin_amdgcn_tensor_load_to_lds(g0, g1, gz4, gz4, gz8, 0);
            __builtin_amdgcn_s_wait_tensorcnt(0);
        }
        __syncthreads();
    }
    // The TDM write is invisible to alias analysis (the LDS address travels inside the
    // descriptor as an integer). ldsOff escaped via ptrtoint above; consume it in an asm
    // barrier with a memory clobber so the staged data is considered written, while
    // keeping Bf itself traceable to LDS address space (=> ds_load_b128, not flat).
    asm volatile("" :: "s"(ldsOff) : "memory");
    const v16bf* Bf = (const v16bf*)ldsB;
#else
    const v16bf* Bf = (const v16bf*)Bt;
#endif

    v8f acc[NT] = {};

    // A-fragment source row (clamped: WMMA needs full EXEC; stores are guarded)
    int arow = rowbase + l15;
    if (arow >= nrows) arow = nrows - 1;
    const float* aptr = A + (size_t)arow * K + half * 16;   // contiguous K layout (matches B)

    for (int kt = 0; kt < ktiles; ++kt) {
        v16bf af;
        const float* pa = aptr + kt * 32;
#pragma unroll
        for (int j = 0; j < 8; ++j) af[j]     = (__bf16)pa[j];        // K = half*16 + j
#pragma unroll
        for (int j = 0; j < 8; ++j) af[j + 8] = (__bf16)pa[j + 8];   // K = half*16 + 8 + j
#pragma unroll
        for (int nt = 0; nt < NT; ++nt) {
            v16bf bf = Bf[(size_t)(kt * NT + nt) * 32 + lane];
            acc[nt] = __builtin_amdgcn_wmma_f32_16x16x32_bf16(
                false, af, false, bf, (short)0, acc[nt], false, false);
        }
    }

    // C layout: VGPR r -> row rowbase + half*8 + r, col = nt*16 + l15
#pragma unroll
    for (int nt = 0; nt < NT; ++nt) {
        const int col = nt * 16 + l15;
        const float b = (flags & 1) ? bias[col] : 0.0f;
#pragma unroll
        for (int r = 0; r < 8; ++r) {
            const int row = rowbase + half * 8 + r;
            if (row < nrows) {
                float v = acc[nt][r] + b;
                if (flags & 2) v = v > 0.f ? v : 0.f;
                C[(size_t)row * HID + col] = v;
            }
        }
    }
}

// ---------- pass 1: raw attention scores + segment max ----------
__global__ __launch_bounds__(256)
void gat_edge_attn(const int* __restrict__ src, const int* __restrict__ dst,
                   const float* __restrict__ eattr,
                   const float* __restrict__ xl, const float* __restrict__ xr,
                   const float* __restrict__ We, const float* __restrict__ att,
                   float* __restrict__ alpha, unsigned* __restrict__ amax, int E_) {
    const int t = blockIdx.x * 256 + threadIdx.x;
    const int e = t >> 7, ch = t & 127;
    if (e >= E_) return;
    const int s = src[e], d = dst[e];
    float m = xl[(size_t)s * HID + ch] + xr[(size_t)d * HID + ch] + eattr[e] * We[ch];
    m = m > 0.f ? m : NEG_SLOPE * m;
    float v = m * att[ch];
#pragma unroll
    for (int off = 8; off; off >>= 1) v += __shfl_xor(v, off, 16);
    if ((threadIdx.x & 15) == 0) {
        const int hh = ch >> 4;
        alpha[(size_t)e * 8 + hh] = v;
        atomicMax(amax + (size_t)d * 8 + hh, f2mono(v));
    }
}

// ---------- pass 2: exp(score - max) + segment sum ----------
__global__ void gat_edge_softmax(const int* __restrict__ dst, float* __restrict__ alpha,
                                 const unsigned* __restrict__ amax,
                                 float* __restrict__ denom, int EH) {
    const int i = blockIdx.x * 256 + threadIdx.x;
    if (i >= EH) return;
    const int e = i >> 3, hh = i & 7;
    const int d = dst[e];
    const float mx = mono2f(amax[(size_t)d * 8 + hh]);
    const float v = __expf(alpha[i] - mx);
    alpha[i] = v;
    atomicAdd(denom + (size_t)d * 8 + hh, v);
}

// ---------- pass 3: weighted scatter of source features ----------
__global__ __launch_bounds__(256)
void gat_edge_aggregate(const int* __restrict__ src, const int* __restrict__ dst,
                        const float* __restrict__ alpha, const float* __restrict__ xl,
                        float* __restrict__ acc, int E_) {
    const int t = blockIdx.x * 256 + threadIdx.x;
    const int e = t >> 7, ch = t & 127;
    if (e >= E_) return;
    const float a = alpha[(size_t)e * 8 + (ch >> 4)];
    const float v = a * xl[(size_t)src[e] * HID + ch];
    atomicAdd(acc + (size_t)dst[e] * HID + ch, v);
}

// ---------- per-node finish: normalize, +bias, relu (in place), pool accumulate ----------
__global__ void gat_node_finish(float* __restrict__ acc, const float* __restrict__ denom,
                                const float* __restrict__ bias, const int* __restrict__ batch,
                                float* __restrict__ pool, int layer, int N_) {
    const int i = blockIdx.x * 256 + threadIdx.x;
    if (i >= N_ * HID) return;
    const int n = i >> 7, ch = i & 127;
    float v = acc[i] / (denom[(size_t)n * 8 + (ch >> 4)] + 1e-16f) + bias[layer * HID + ch];
    v = v > 0.f ? v : 0.f;
    acc[i] = v;                                   // becomes h for next layer
    atomicAdd(pool + (size_t)batch[n] * 384 + layer * HID + ch, v);
}

__global__ void gat_count(const int* __restrict__ batch, float* __restrict__ cnt, int N_) {
    const int i = blockIdx.x * 256 + threadIdx.x;
    if (i < N_) atomicAdd(cnt + batch[i], 1.0f);
}

__global__ void gat_final(const float* __restrict__ pool, const float* __restrict__ cnt,
                          float* __restrict__ out, int total) {
    const int i = blockIdx.x * 256 + threadIdx.x;
    if (i >= total) return;
    float c = cnt[i / 384];
    c = c > 1.f ? c : 1.f;
    out[i] = pool[i] / c;
}

extern "C" void kernel_launch(void* const* d_in, const int* in_sizes, int n_in,
                              void* d_out, int out_size, void* d_ws, size_t ws_size,
                              hipStream_t stream) {
    const float* x       = (const float*)d_in[0];
    const int*   ei      = (const int*)  d_in[1];
    const float* eattr   = (const float*)d_in[2];
    const int*   batch   = (const int*)  d_in[3];
    const float* w_pre1  = (const float*)d_in[4];
    const float* b_pre1  = (const float*)d_in[5];
    const float* w_pre2  = (const float*)d_in[6];
    const float* b_pre2  = (const float*)d_in[7];
    const float* Wl      = (const float*)d_in[8];
    const float* Wr      = (const float*)d_in[9];
    const float* We      = (const float*)d_in[10];
    const float* att     = (const float*)d_in[11];
    const float* biasc   = (const float*)d_in[12];

    const int N    = in_sizes[3];
    const int F_IN = in_sizes[0] / N;
    const int E    = in_sizes[2];
    const int G    = out_size / (3 * HID);
    const int* src = ei;
    const int* dst = ei + E;

    // ---- workspace carve-out (256 B aligned) ----
    size_t off = 0;
    auto carve = [&](size_t bytes) {
        void* p = (char*)d_ws + off;
        off += (bytes + 255) & ~(size_t)255;
        return p;
    };
    float*    hA     = (float*)   carve((size_t)N * HID * 4);   // ping
    float*    hB     = (float*)   carve((size_t)N * HID * 4);   // pong
    float*    xl     = (float*)   carve((size_t)N * HID * 4);
    float*    xr     = (float*)   carve((size_t)N * HID * 4);
    float*    alpha  = (float*)   carve((size_t)E * 8 * 4);
    unsigned* amax   = (unsigned*)carve((size_t)N * 8 * 4);
    float*    denom  = (float*)   carve((size_t)N * 8 * 4);
    float*    pool   = (float*)   carve((size_t)G * 384 * 4);
    float*    cnt    = (float*)   carve((size_t)G * 4);
    __bf16*   wbpre1 = (__bf16*)  carve((size_t)F_IN * HID * 2);
    __bf16*   wbpre2 = (__bf16*)  carve((size_t)HID * HID * 2);
    __bf16*   wbl    = (__bf16*)  carve((size_t)3 * HID * HID * 2);
    __bf16*   wbr    = (__bf16*)  carve((size_t)3 * HID * HID * 2);
    (void)ws_size; (void)n_in;

    // ---- one-time per call inits ----
    (void)hipMemsetAsync(pool, 0, (size_t)G * 384 * 4, stream);
    (void)hipMemsetAsync(cnt,  0, (size_t)G * 4,       stream);
    gat_count<<<(N + 255) / 256, 256, 0, stream>>>(batch, cnt, N);

    // ---- weight swizzle (bf16 WMMA B-fragment layout) ----
    gat_prep_b<<<(F_IN / 32) * NT, 32, 0, stream>>>(w_pre1, wbpre1);
    gat_prep_b<<<(HID  / 32) * NT, 32, 0, stream>>>(w_pre2, wbpre2);
    for (int l = 0; l < 3; ++l) {
        gat_prep_b<<<(HID / 32) * NT, 32, 0, stream>>>(Wl + (size_t)l * HID * HID, wbl + (size_t)l * HID * HID);
        gat_prep_b<<<(HID / 32) * NT, 32, 0, stream>>>(Wr + (size_t)l * HID * HID, wbr + (size_t)l * HID * HID);
    }

    const int gemmGrid = (N + 63) / 64;
    // pre-MLP
    gat_gemm_wmma<<<gemmGrid, 128, 0, stream>>>(x,  wbpre1, b_pre1, hA, N, F_IN, 3);
    gat_gemm_wmma<<<gemmGrid, 128, 0, stream>>>(hA, wbpre2, b_pre2, hB, N, HID,  3);

    float* h   = hB;
    float* acc = hA;
    const int edgeGrid = (E * 128 + 255) / 256;
    for (int l = 0; l < 3; ++l) {
        gat_gemm_wmma<<<gemmGrid, 128, 0, stream>>>(h, wbl + (size_t)l * HID * HID, nullptr, xl, N, HID, 0);
        gat_gemm_wmma<<<gemmGrid, 128, 0, stream>>>(h, wbr + (size_t)l * HID * HID, nullptr, xr, N, HID, 0);

        (void)hipMemsetAsync(amax,  0, (size_t)N * 8 * 4,   stream);  // mono(0) < all reals
        (void)hipMemsetAsync(denom, 0, (size_t)N * 8 * 4,   stream);
        (void)hipMemsetAsync(acc,   0, (size_t)N * HID * 4, stream);

        gat_edge_attn<<<edgeGrid, 256, 0, stream>>>(src, dst, eattr, xl, xr,
                                                    We + l * HID, att + l * HID,
                                                    alpha, amax, E);
        gat_edge_softmax<<<(E * 8 + 255) / 256, 256, 0, stream>>>(dst, alpha, amax, denom, E * 8);
        gat_edge_aggregate<<<edgeGrid, 256, 0, stream>>>(src, dst, alpha, xl, acc, E);
        gat_node_finish<<<(N * HID + 255) / 256, 256, 0, stream>>>(acc, denom, biasc, batch, pool, l, N);

        float* t = h; h = acc; acc = t;            // acc became new h
    }

    gat_final<<<(G * 384 + 255) / 256, 256, 0, stream>>>(pool, cnt, (float*)d_out, G * 384);
}
